// GraphCN_18854906429735
// MI455X (gfx1250) — matrix operand order repaired and checked by reading, not compiled, not run on hardware
//
#include <hip/hip_runtime.h>

// ---------------------------------------------------------------------------
// GCN (4 layers) on gfx1250 (MI455X, wave32, WMMA).
//  - GEMMs via v_wmma_f32_16x16x32_bf16 (fp32 accumulate), inputs RNE-rounded
//    to bf16 into a zero-padded workspace (so EXEC is all-1s in WMMA waves).
//  - Edge aggregation: one wave per edge, lane f handles features f, f+32,
//    native fp32 global atomics (L2-resident: h is 25.6MB << 192MB L2).
// ---------------------------------------------------------------------------

typedef __attribute__((ext_vector_type(16))) __bf16 v16bf;
typedef __attribute__((ext_vector_type(8)))  float  v8f;
typedef __attribute__((ext_vector_type(4)))  unsigned int u32x4;

union Frag16 { u32x4 u[2]; v16bf v; };

__device__ __forceinline__ unsigned short f2bf(float f) {
  // round-to-nearest-even f32 -> bf16
  unsigned int u = __float_as_uint(f);
  u += 0x7fffu + ((u >> 16) & 1u);
  return (unsigned short)(u >> 16);
}

__device__ __forceinline__ void atomAddF(float* p, float v) {
  (void)__hip_atomic_fetch_add(p, v, __ATOMIC_RELAXED, __HIP_MEMORY_SCOPE_AGENT);
}

// ------------------------------ prep kernels -------------------------------

__global__ __launch_bounds__(256) void k_fill_ones(float* deg, int n) {
  int i = blockIdx.x * blockDim.x + threadIdx.x;
  if (i < n) deg[i] = 1.0f;                       // self-loop contributes 1
}

__global__ __launch_bounds__(256) void k_deg_count(const int* col, float* deg, int E) {
  int i = blockIdx.x * blockDim.x + threadIdx.x;
  if (i < E) atomAddF(&deg[col[i]], 1.0f);
}

__global__ __launch_bounds__(256) void k_rsqrt_inplace(float* d, int n) {
  int i = blockIdx.x * blockDim.x + threadIdx.x;
  if (i < n) d[i] = rsqrtf(d[i]);                 // deg >= 1 always
}

__global__ __launch_bounds__(256) void k_norm(const int* ei, const float* dis,
                                              float* nrm, int E, int n) {
  int i = blockIdx.x * blockDim.x + threadIdx.x;
  if (i < E) {
    nrm[i] = dis[ei[i]] * dis[ei[E + i]];
  } else if (i < E + n) {
    float v = dis[i - E];
    nrm[i] = v * v;                               // self-loop: 1/deg
  }
}

// input (n x Kin, f32) -> zero-padded bf16 (n x 2^kshift), optional fused relu
__global__ __launch_bounds__(256) void k_cvt_x(const float* __restrict__ in,
                                               unsigned short* __restrict__ xb,
                                               int n, int Kin, int kshift, int relu) {
  long long t = (long long)blockIdx.x * blockDim.x + threadIdx.x;
  if (t >= ((long long)n << kshift)) return;
  int Kpad = 1 << kshift;
  int k = (int)(t & (Kpad - 1));
  long long r = t >> kshift;
  float v = (k < Kin) ? in[r * Kin + k] : 0.0f;
  if (relu) v = fmaxf(v, 0.0f);
  xb[t] = f2bf(v);
}

// W (Kin x OUT, f32, row-major) -> Wt (OUTpad x Kpad, bf16, column-major in K)
__global__ __launch_bounds__(256) void k_cvt_w(const float* __restrict__ W,
                                               unsigned short* __restrict__ wt,
                                               int Kin, int Kpad, int OUT, int OUTpad) {
  int t = blockIdx.x * blockDim.x + threadIdx.x;
  if (t >= Kpad * OUTpad) return;
  int k = t % Kpad;
  int o = t / Kpad;
  float v = (k < Kin && o < OUT) ? W[k * OUT + o] : 0.0f;
  wt[t] = f2bf(v);
}

// ---------------------------- WMMA GEMM ------------------------------------
// One wave per 16-row M tile; per tile: OUTpad/16 accumulators of 16x16,
// K loop in chunks of 32 with v_wmma_f32_16x16x32_bf16.
__global__ __launch_bounds__(256) void k_gemm_wmma(const unsigned short* __restrict__ xb,
                                                   const unsigned short* __restrict__ wt,
                                                   float* __restrict__ h,
                                                   int mTiles, int Kpad, int OUTpad) {
  int wave = blockIdx.x * 8 + (threadIdx.x >> 5);  // 256 threads = 8 waves
  if (wave >= mTiles) return;                      // wave-uniform: EXEC stays full
  int lane = threadIdx.x & 31;
  int lo15 = lane & 15;
  int hi   = lane >> 4;

  int mBase = wave << 4;
  const unsigned short* arow = xb + (size_t)(mBase + lo15) * Kpad;

  int nT = OUTpad >> 4;
  for (int nt = 0; nt < nT; ++nt) {
    v8f acc = {};
    // B fragment: column nt*16 + lo15, K = kc + hi*8 + {0..7}, kc + 16 + hi*8 + {0..7}
    const unsigned short* bcol = wt + (size_t)((nt << 4) + lo15) * Kpad + (hi << 3);
    for (int kc = 0; kc < Kpad; kc += 32) {
      Frag16 a, b;
      // A layout (ISA 16-bit A 16x32): elems 0..7 -> K = kc+hi*8+{0..7},
      //                                elems 8..15 -> K = kc+16+hi*8+{0..7}
      a.u[0] = *(const u32x4*)(arow + kc + hi * 8);
      a.u[1] = *(const u32x4*)(arow + kc + hi * 8 + 16);
      b.u[0] = *(const u32x4*)(bcol + kc);
      b.u[1] = *(const u32x4*)(bcol + kc + 16);
      acc = __builtin_amdgcn_wmma_f32_16x16x32_bf16(
          /*neg_a=*/false, a.v, /*neg_b=*/false, b.v,
          /*c_mod=*/(short)0, acc, /*reuse_a=*/false, /*reuse_b=*/false);
    }
    // C/D layout: VGPR v -> row mBase + hi*8 + v, col nt*16 + lo15
    float* o = h + (size_t)(mBase + hi * 8) * OUTpad + (nt << 4) + lo15;
#pragma unroll
    for (int v = 0; v < 8; ++v) o[(size_t)v * OUTpad] = acc[v];
  }
}

// out[n x F] = bias broadcast (aggregation target initializer)
__global__ __launch_bounds__(256) void k_bias_fill(float* __restrict__ out,
                                                   const float* __restrict__ b,
                                                   int n, int F) {
  long long t = (long long)blockIdx.x * blockDim.x + threadIdx.x;
  if (t >= (long long)n * F) return;
  out[t] = b[(int)(t % F)];
}

// scatter-add: one wave per edge (incl. self loops), lane covers f, f+32
__global__ __launch_bounds__(256) void k_aggregate(const float* __restrict__ h, int Hs,
                                                   const int* __restrict__ ei,
                                                   const float* __restrict__ nrm,
                                                   float* __restrict__ out, int Os,
                                                   int F, int E, int n) {
  long long t = (long long)blockIdx.x * blockDim.x + threadIdx.x;
  int e = (int)(t >> 5);
  if (e >= E + n) return;
  int lane = (int)(t & 31);
  int r, c;
  if (e < E) { r = ei[e]; c = ei[E + e]; }
  else       { r = e - E; c = r; }                 // self loop
  float w = nrm[e];
  const float* hrow = h + (size_t)r * Hs;
  float* orow = out + (size_t)c * Os;
  for (int f = lane; f < F; f += 32) atomAddF(orow + f, hrow[f] * w);
}

// ------------------------------ host side ----------------------------------

extern "C" void kernel_launch(void* const* d_in, const int* in_sizes, int n_in,
                              void* d_out, int out_size, void* d_ws, size_t ws_size,
                              hipStream_t stream) {
  const float* x  = (const float*)d_in[0];
  const int*   ei = (const int*)d_in[1];           // (2, E): row then col
  const float* W1 = (const float*)d_in[2];  const float* b1 = (const float*)d_in[3];
  const float* W2 = (const float*)d_in[4];  const float* b2 = (const float*)d_in[5];
  const float* W3 = (const float*)d_in[6];  const float* b3 = (const float*)d_in[7];
  const float* Wo = (const float*)d_in[8];  const float* bo = (const float*)d_in[9];

  const int E = in_sizes[1] / 2;
  const int n = in_sizes[0] / 100;                 // 100000 (divisible by 16)

  // workspace carve-out (256B aligned)
  char* ws = (char*)d_ws;
  auto al = [](size_t v) { return (v + 255) & ~(size_t)255; };
  size_t off = 0;
  float* dis          = (float*)(ws + off);          off = al(off + (size_t)n * 4);
  float* nrm          = (float*)(ws + off);          off = al(off + (size_t)(E + n) * 4);
  unsigned short* xb  = (unsigned short*)(ws + off); off = al(off + (size_t)n * 128 * 2);
  float* H            = (float*)(ws + off);          off = al(off + (size_t)n * 64 * 4);
  float* bufA         = (float*)(ws + off);          off = al(off + (size_t)n * 64 * 4);
  unsigned short* wt  = (unsigned short*)(ws + off); off = al(off + (size_t)128 * 64 * 2);
  (void)ws_size;

  const int B = 256;
  // ---- degree / norm prep ----
  k_fill_ones    <<<(n + B - 1) / B, B, 0, stream>>>(dis, n);
  k_deg_count    <<<(E + B - 1) / B, B, 0, stream>>>(ei + E, dis, E);
  k_rsqrt_inplace<<<(n + B - 1) / B, B, 0, stream>>>(dis, n);
  k_norm         <<<(E + n + B - 1) / B, B, 0, stream>>>(ei, dis, nrm, E, n);

  const int mTiles = n / 16;
  const int gemmBlocks = (mTiles + 7) / 8;
  const long long aggT = 32LL * (E + n);

  auto layer = [&](const float* in, int Kin, int kshift, int relu,
                   const float* W, int OUT, int OUTpad,
                   const float* bias, float* out, int Os) {
    const int Kpad = 1 << kshift;
    const long long cvtN = (long long)n << kshift;
    k_cvt_x<<<(unsigned)((cvtN + B - 1) / B), B, 0, stream>>>(in, xb, n, Kin, kshift, relu);
    k_cvt_w<<<(Kpad * OUTpad + B - 1) / B, B, 0, stream>>>(W, wt, Kin, Kpad, OUT, OUTpad);
    k_gemm_wmma<<<gemmBlocks, B, 0, stream>>>(xb, wt, H, mTiles, Kpad, OUTpad);
    const long long bN = (long long)n * Os;
    k_bias_fill<<<(unsigned)((bN + B - 1) / B), B, 0, stream>>>(out, bias, n, Os);
    k_aggregate<<<(unsigned)((aggT + B - 1) / B), B, 0, stream>>>(H, OUTpad, ei, nrm,
                                                                  out, Os, OUT, E, n);
  };

  // L1: x(100) -> 64, no input relu; L2/L3: 64 -> 64 with fused relu;
  // L4: 64 -> 18 (padded to 32 inside GEMM), writes d_out directly.
  layer(x,    100, 7, 0, W1, 64, 64, b1, bufA, 64);
  layer(bufA,  64, 6, 1, W2, 64, 64, b2, bufA, 64);
  layer(bufA,  64, 6, 1, W3, 64, 64, b3, bufA, 64);
  layer(bufA,  64, 6, 1, Wo, 18, 32, bo, (float*)d_out, 18);
}
